// Attention_72894184947779
// MI455X (gfx1250) — matrix operand order repaired and checked by reading, not compiled, not run on hardware
//
#include <hip/hip_runtime.h>
#include <hip/hip_bf16.h>
#include <stdint.h>

typedef __attribute__((ext_vector_type(16))) _Float16 v16h;
typedef __attribute__((ext_vector_type(8)))  _Float16 v8h;
typedef __attribute__((ext_vector_type(8)))  float    v8f;
typedef __attribute__((ext_vector_type(4)))  float    v4f;

#define IN_CH 256
#define EMBED 128
#define HWSZ  4096
#define OQKV  384   // 3*EMBED
#define NB    4

static __device__ __forceinline__ v8f zero8() {
  v8f z;
#pragma unroll
  for (int t = 0; t < 8; ++t) z[t] = 0.0f;
  return z;
}

static __device__ __forceinline__ v16h pack16(v8h lo, v8h hi) {
  v16h a;
#pragma unroll
  for (int t = 0; t < 8; ++t) { a[t] = lo[t]; a[8 + t] = hi[t]; }
  return a;
}

// A fragment (16x32 f16, MxK), row-major source, ld in halves.
// lane L: row m = L&15; K(h) = (h&7) + 8*(L>=16) + 16*(h>=8)  -> two contiguous 8-half runs.
static __device__ __forceinline__ v16h load_a_h(const _Float16* base, int ld, int lane) {
  const int m  = lane & 15;
  const int ko = (lane & 16) ? 8 : 0;
  const _Float16* r = base + (size_t)m * ld;
  v8h lo = *(const v8h*)(r + ko);
  v8h hi = *(const v8h*)(r + 16 + ko);
  return pack16(lo, hi);
}

// Same A fragment but converting from f32 source on the fly.
static __device__ __forceinline__ v16h load_a_f32(const float* base, int ld, int lane) {
  const int m  = lane & 15;
  const int ko = (lane & 16) ? 8 : 0;
  const float* r = base + (size_t)m * ld;
  v4f x0 = *(const v4f*)(r + ko);
  v4f x1 = *(const v4f*)(r + ko + 4);
  v4f x2 = *(const v4f*)(r + 16 + ko);
  v4f x3 = *(const v4f*)(r + 16 + ko + 4);
  v16h a;
#pragma unroll
  for (int t = 0; t < 4; ++t) {
    a[t]      = (_Float16)x0[t];
    a[4 + t]  = (_Float16)x1[t];
    a[8 + t]  = (_Float16)x2[t];
    a[12 + t] = (_Float16)x3[t];
  }
  return a;
}

// B fragment (32x16 f16, KxN) loaded from a TRANSPOSED buffer Bt[n][k] (ld in halves).
// lane L: column n = L&15; K(h) = (h&7) + 8*(L>=16) + 16*(h>=8) -> two contiguous 8-half runs.
static __device__ __forceinline__ v16h load_bt_h(const _Float16* base, int ld, int lane) {
  const int n  = lane & 15;
  const int ko = (lane & 16) ? 8 : 0;
  const _Float16* r = base + (size_t)n * ld + ko;
  v8h lo = *(const v8h*)(r);
  v8h hi = *(const v8h*)(r + 16);
  return pack16(lo, hi);
}

static __device__ __forceinline__ v8f wmma16(v16h a, v16h b, v8f c) {
  return __builtin_amdgcn_wmma_f32_16x16x32_f16(false, a, false, b, (short)0, c, false, false);
}

// ---------------------------------------------------------------------------
// Kernel 1: QKV = W_qkv(384x256) @ X(256x4096) + b, per batch.
// Writes qkv f16 in [o][p] flat order (so the reshape views q/k/v as
// row-major with stride 384), plus a scattered transposed copy of V: vT[d][i].
// ---------------------------------------------------------------------------
__global__ __launch_bounds__(256)
void k_qkv(const float* __restrict__ x, const float* __restrict__ Wqkv,
           const float* __restrict__ bqkv, _Float16* __restrict__ qkv,
           _Float16* __restrict__ vT)
{
  const int tid  = threadIdx.x;
  const int lane = tid & 31;
  const int w    = tid >> 5;   // 0..7
  const int wm   = w & 3;      // 4 m-tiles of 16 -> 64 rows
  const int wn   = w >> 2;     // 2 n-halves of 64
  const int p0   = blockIdx.x * 128;
  const int o0   = blockIdx.y * 64;
  const int nb   = blockIdx.z;

  const float* xb = x + (size_t)nb * IN_CH * HWSZ;
  _Float16* qkvb  = qkv + (size_t)nb * OQKV * HWSZ;
  _Float16* vTb   = vT  + (size_t)nb * EMBED * HWSZ;

  __shared__ __align__(16) _Float16 xT[128 * 40];  // [p][c], padded row stride 40 (80B, 16B-aligned)

  v8f acc[4];
#pragma unroll
  for (int t = 0; t < 4; ++t) acc[t] = zero8();

  for (int k0 = 0; k0 < IN_CH; k0 += 32) {
    // stage x[k0..k0+31][p0..p0+127] into LDS transposed as f16
#pragma unroll
    for (int r = 0; r < 16; ++r) {
      int idx = tid + 256 * r;
      int c = idx >> 7;        // 0..31
      int p = idx & 127;
      xT[p * 40 + c] = (_Float16)xb[(size_t)(k0 + c) * HWSZ + p0 + p];
    }
    __syncthreads();

    v16h a = load_a_f32(Wqkv + (size_t)(o0 + 16 * wm) * IN_CH + k0, IN_CH, lane);
#pragma unroll
    for (int t = 0; t < 4; ++t) {
      v16h b = load_bt_h(&xT[(wn * 64 + 16 * t) * 40], 40, lane);
      acc[t] = wmma16(a, b, acc[t]);
    }
    __syncthreads();
  }

#pragma unroll
  for (int t = 0; t < 4; ++t) {
#pragma unroll
    for (int r = 0; r < 8; ++r) {
      int o = o0 + 16 * wm + r + ((lane & 16) ? 8 : 0);
      int p = p0 + 64 * wn + 16 * t + (lane & 15);
      float v = acc[t][r] + bqkv[o];
      unsigned F = (unsigned)o * HWSZ + (unsigned)p;   // flat index within batch
      _Float16 hv = (_Float16)v;
      qkvb[F] = hv;
      unsigned i  = F / 384u;                           // token index in reshaped view
      unsigned dd = F - i * 384u;                       // channel in [0,384)
      if (dd >= 256u) vTb[(size_t)(dd - 256u) * HWSZ + i] = hv;  // V transposed copy
    }
  }
}

// ---------------------------------------------------------------------------
// Kernel 2: per-column online softmax stats over s[i,j] = (q_i . k_j)/64.
// Block: 16 columns j, 4 waves split the 4096 i's; online (max,sum) per lane,
// tree-combined through LDS. Prefetches the next Q tile (global_prefetch_b8).
// ---------------------------------------------------------------------------
__global__ __launch_bounds__(128)
void k_colstats(const _Float16* __restrict__ qkv, float* __restrict__ colmax,
                float* __restrict__ colsum)
{
  const int tid = threadIdx.x, lane = tid & 31, w = tid >> 5;
  const int j0 = blockIdx.x * 16;
  const int nb = blockIdx.y;
  const _Float16* qkvb = qkv + (size_t)nb * OQKV * HWSZ;

  // K rows j0..j0+15 are fixed for the block: preload the 4 B fragments (d = 0..127)
  v16h bk[4];
#pragma unroll
  for (int kk = 0; kk < 4; ++kk)
    bk[kk] = load_bt_h(qkvb + (size_t)j0 * OQKV + EMBED + 32 * kk, OQKV, lane);

  float m_run = -3.0e38f, s_run = 0.0f;

  for (int it = 0; it < 64; ++it) {
    const int i0 = (w + 4 * it) * 16;
    // prefetch next i-tile's q rows into cache (speculative; tail lands in ws)
    __builtin_prefetch(qkvb + (size_t)(i0 + 64 + (lane & 15)) * OQKV, 0, 1);
    v8f s = zero8();
#pragma unroll
    for (int kk = 0; kk < 4; ++kk) {
      v16h a = load_a_h(qkvb + (size_t)i0 * OQKV + 32 * kk, OQKV, lane);
      s = wmma16(a, bk[kk], s);
    }
    float sv[8];
    float vmax = -3.0e38f;
#pragma unroll
    for (int r = 0; r < 8; ++r) { sv[r] = s[r] * 0.015625f; vmax = fmaxf(vmax, sv[r]); }
    float nm  = fmaxf(m_run, vmax);
    float acc = s_run * __expf(m_run - nm);
#pragma unroll
    for (int r = 0; r < 8; ++r) acc += __expf(sv[r] - nm);
    m_run = nm; s_run = acc;
  }

  __shared__ float redm[4][32], reds[4][32];
  redm[w][lane] = m_run; reds[w][lane] = s_run;
  __syncthreads();
  if (tid < 16) {     // one thread per column; combine 4 waves x 2 lane-halves
    float M = -3.0e38f;
#pragma unroll
    for (int ww = 0; ww < 4; ++ww)
      M = fmaxf(M, fmaxf(redm[ww][tid], redm[ww][tid + 16]));
    float S = 0.0f;
#pragma unroll
    for (int ww = 0; ww < 4; ++ww) {
      S += reds[ww][tid]      * __expf(redm[ww][tid]      - M);
      S += reds[ww][tid + 16] * __expf(redm[ww][tid + 16] - M);
    }
    colmax[nb * HWSZ + j0 + tid] = M;
    colsum[nb * HWSZ + j0 + tid] = S;
  }
}

// ---------------------------------------------------------------------------
// Kernel 3: out[i,d] = sum_j exp(s[i,j]-colmax[j])/colsum[j] * v[j,d], fused.
// Cooperative version: per 128-j outer iteration,
//   1) all lanes kick off an async DMA of the 128x128 vT tile into LDS
//      (global_load_async_to_lds_b128, ASYNCcnt) -- overlaps with (2);
//   2) wave w computes ONLY its 16x32 P chunk (j in [J0+32w, J0+32w+32)) via
//      QK^T WMMA + column-softmax renorm, publishes it to LDS;
//   3) after s_wait_asynccnt + barrier, every wave consumes all four P chunks
//      against its own 32 d-columns (8 P.V WMMAs) from the LDS V tile.
// This removes the 4x redundant QK^T recompute of the naive scheme.
// Result scattered into O2^T[p][e].
// ---------------------------------------------------------------------------
__global__ __launch_bounds__(128)
void k_attn_pv(const _Float16* __restrict__ qkv, const _Float16* __restrict__ vT,
               const float* __restrict__ colmax, const float* __restrict__ colsum,
               _Float16* __restrict__ o2t)
{
  const int tid = threadIdx.x, lane = tid & 31, w = tid >> 5;
  const int i0 = blockIdx.x * 16;
  const int nb = blockIdx.y;
  const _Float16* qkvb = qkv + (size_t)nb * OQKV * HWSZ;
  const _Float16* vTb  = vT  + (size_t)nb * EMBED * HWSZ;
  const float* cmb = colmax + nb * HWSZ;
  const float* csb = colsum + nb * HWSZ;
  _Float16* o2b = o2t + (size_t)nb * EMBED * HWSZ;

  __shared__ __align__(16) _Float16 pbuf[4][16 * 40];  // P chunks [m][j], padded rows (80B)
  __shared__ __align__(16) _Float16 vbuf[128 * 136];   // vT tile [d][j], padded rows (272B)

  const unsigned vbase = (unsigned)(uintptr_t)(&vbuf[0]);

  v16h aq[4];
#pragma unroll
  for (int kk = 0; kk < 4; ++kk)
    aq[kk] = load_a_h(qkvb + (size_t)i0 * OQKV + 32 * kk, OQKV, lane);

  v8f acc0 = zero8(), acc1 = zero8();
  const int dw = 32 * w;
  const int mrow0 = (lane & 16) ? 8 : 0;

  for (int jo = 0; jo < 32; ++jo) {
    const int J0 = 128 * jo;

    // (1) async-copy vT[0:128][J0:J0+128] -> vbuf (2048 x b128, 16 per thread)
#pragma unroll
    for (int r = 0; r < 16; ++r) {
      int idx = tid + 128 * r;
      int row = idx >> 4;       // 0..127 (d)
      int seg = idx & 15;       // 16B segment within row
      const _Float16* g = vTb + (size_t)row * HWSZ + J0 + seg * 8;
      unsigned l = vbase + (unsigned)(row * 272 + seg * 16);
      asm volatile("global_load_async_to_lds_b128 %0, %1, off"
                   :: "v"(l), "v"(g) : "memory");
    }

    // (2) this wave's P chunk: j in [J0+32w, J0+32w+32)
    const int js0 = J0 + 32 * w;
#pragma unroll
    for (int sub = 0; sub < 2; ++sub) {
      const int js = js0 + 16 * sub;
      v8f s = zero8();
#pragma unroll
      for (int kk = 0; kk < 4; ++kk) {
        v16h bkf = load_bt_h(qkvb + (size_t)js * OQKV + EMBED + 32 * kk, OQKV, lane);
        s = wmma16(aq[kk], bkf, s);
      }
      const int jcol = js + (lane & 15);
      const float cm  = cmb[jcol];
      const float rcs = 1.0f / csb[jcol];
#pragma unroll
      for (int r = 0; r < 8; ++r) {
        float pv = __expf(s[r] * 0.015625f - cm) * rcs;
        pbuf[w][(mrow0 + r) * 40 + 16 * sub + (lane & 15)] = (_Float16)pv;
      }
    }

    asm volatile("s_wait_asynccnt 0" ::: "memory");
    __syncthreads();

    // (3) consume all four P chunks against this wave's 32 d-columns
#pragma unroll
    for (int wc = 0; wc < 4; ++wc) {
      v16h ap = load_a_h(&pbuf[wc][0], 40, lane);
      v16h b0 = load_bt_h(&vbuf[dw * 136 + 32 * wc], 136, lane);
      v16h b1 = load_bt_h(&vbuf[(dw + 16) * 136 + 32 * wc], 136, lane);
      acc0 = wmma16(ap, b0, acc0);
      acc1 = wmma16(ap, b1, acc1);
    }
    __syncthreads();
  }

  // scatter out[i,d] into O2^T[p][e]  (F = i*128+d ; e = F>>12 ; p = F&4095)
#pragma unroll
  for (int t = 0; t < 2; ++t) {
#pragma unroll
    for (int r = 0; r < 8; ++r) {
      int i = i0 + r + mrow0;
      int d = dw + 16 * t + (lane & 15);
      unsigned F = (unsigned)i * EMBED + (unsigned)d;
      unsigned e = F >> 12;
      unsigned p = F & 4095u;
      float v = (t == 0) ? acc0[r] : acc1[r];
      o2b[(size_t)p * EMBED + e] = (_Float16)v;
    }
  }
}

// ---------------------------------------------------------------------------
// Kernel 4: y = W_out(256x128) @ O2(128x4096) + b_out.  B operand comes from
// the pre-transposed O2^T buffer, so both fragments load straight from global.
// ---------------------------------------------------------------------------
__global__ __launch_bounds__(256)
void k_proj(const _Float16* __restrict__ o2t, const float* __restrict__ Wout,
            const float* __restrict__ bout, float* __restrict__ y)
{
  const int tid = threadIdx.x, lane = tid & 31, w = tid >> 5;
  const int p0 = blockIdx.x * 64;
  const int c0 = blockIdx.y * 128 + 16 * w;
  const int nb = blockIdx.z;
  const _Float16* o2b = o2t + (size_t)nb * EMBED * HWSZ;

  v8f acc[4];
#pragma unroll
  for (int t = 0; t < 4; ++t) acc[t] = zero8();

#pragma unroll
  for (int kk = 0; kk < 4; ++kk) {
    v16h a = load_a_f32(Wout + (size_t)c0 * EMBED + 32 * kk, EMBED, lane);
#pragma unroll
    for (int t = 0; t < 4; ++t) {
      v16h b = load_bt_h(o2b + (size_t)(p0 + 16 * t) * EMBED + 32 * kk, EMBED, lane);
      acc[t] = wmma16(a, b, acc[t]);
    }
  }

#pragma unroll
  for (int t = 0; t < 4; ++t) {
#pragma unroll
    for (int r = 0; r < 8; ++r) {
      int c = c0 + r + ((lane & 16) ? 8 : 0);
      int p = p0 + 16 * t + (lane & 15);
      y[(size_t)nb * IN_CH * HWSZ + (size_t)c * HWSZ + p] = acc[t][r] + bout[c];
    }
  }
}

// ---------------------------------------------------------------------------
extern "C" void kernel_launch(void* const* d_in, const int* in_sizes, int n_in,
                              void* d_out, int out_size, void* d_ws, size_t ws_size,
                              hipStream_t stream) {
  (void)in_sizes; (void)n_in; (void)out_size; (void)ws_size;
  const float* x    = (const float*)d_in[0];
  const float* Wqkv = (const float*)d_in[1];
  const float* bqkv = (const float*)d_in[2];
  const float* Wout = (const float*)d_in[3];
  const float* bout = (const float*)d_in[4];
  float* y = (float*)d_out;

  // workspace layout (~20.1 MB total, all 16B-aligned)
  _Float16* qkv  = (_Float16*)d_ws;                         // 4*384*4096 f16
  _Float16* vT   = qkv + (size_t)NB * OQKV * HWSZ;          // 4*128*4096 f16
  float*    cmax = (float*)(vT + (size_t)NB * EMBED * HWSZ);// 4*4096 f32
  float*    csum = cmax + NB * HWSZ;                        // 4*4096 f32
  _Float16* o2t  = (_Float16*)(csum + NB * HWSZ);           // 4*4096*128 f16

  k_qkv     <<<dim3(32, 6, 4),  256, 0, stream>>>(x, Wqkv, bqkv, qkv, vT);
  k_colstats<<<dim3(256, 4),    128, 0, stream>>>(qkv, cmax, csum);
  k_attn_pv <<<dim3(256, 4),    128, 0, stream>>>(qkv, vT, cmax, csum, o2t);
  k_proj    <<<dim3(64, 2, 4),  256, 0, stream>>>(o2t, Wout, bout, y);
}
